// RCNN_18210661335689
// MI455X (gfx1250) — matrix operand order repaired and checked by reading, not compiled, not run on hardware
//
#include <hip/hip_runtime.h>

// Problem dims (match the reference)
#define Bv 128
#define Tv 1024
#define Dv 512
#define Hv 512

typedef __attribute__((ext_vector_type(16))) __bf16        v16bf;
typedef __attribute__((ext_vector_type(8)))  float         v8f;
typedef __attribute__((ext_vector_type(8)))  unsigned int  v8u;

// ---- helpers -------------------------------------------------------------

__device__ __forceinline__ unsigned short f2bf(float f) {
    unsigned u = __builtin_bit_cast(unsigned, f);
    u += 0x7fffu + ((u >> 16) & 1u);   // round-to-nearest-even
    return (unsigned short)(u >> 16);
}

__device__ __forceinline__ unsigned int packbf(float lo, float hi) {
    return (unsigned int)f2bf(lo) | ((unsigned int)f2bf(hi) << 16);
}

__device__ __forceinline__ v16bf frag_from(uint4 lo, uint4 hi) {
    v8u u = (v8u){lo.x, lo.y, lo.z, lo.w, hi.x, hi.y, hi.z, hi.w};
    return __builtin_bit_cast(v16bf, u);
}

// A-matrix (16x32 bf16, row-major in LDS, row stride 512 elems).
// Lane l: row = l%16; lanes<16 hold K=[0..15], lanes>=16 hold K=[16..31].
__device__ __forceinline__ v16bf load_a_lds(const unsigned short* s, int row, int k0, int half) {
    const uint4* p0 = (const uint4*)(s + row * 512 + k0 + half * 16);
    const uint4* p1 = (const uint4*)(s + row * 512 + k0 + 8 + half * 16);
    return frag_from(*p0, *p1);
}

// B-matrix fragment pre-packed as [lane][8 dwords] (1KB per 32x16 tile).
__device__ __forceinline__ v16bf load_b_glb(const unsigned int* p) {
    const uint4* q = (const uint4*)p;
    return frag_from(q[0], q[1]);
}

__device__ __forceinline__ v8f wmma_bf16(v16bf a, v16bf b, v8f c) {
    return __builtin_amdgcn_wmma_f32_16x16x32_bf16(false, a, false, b, (short)0, c, false, false);
}

// ---- weight packer -------------------------------------------------------
// Packs W (rows = output features n, cols = K=512, fp32) into bf16 B-fragment
// tiles: frag(nt,kt) holds B[k][n] = W[n][k] for k in [kt*32, kt*32+32),
// n in [nt*16, nt*16+16), laid out [lane][8 dwords]:
//   lane<16 : n = nt*16+lane,    dword v = pack(W[n][kt*32+2v],   W[n][kt*32+2v+1])
//   lane>=16: n = nt*16+lane-16, dword v = pack(W[n][kt*32+16+2v],W[n][kt*32+17+2v])
// blocks 0..511   -> W_hh (nt 0..31, kt 0..15) -> whh_p
// blocks 512..2047-> W_ih (nt 0..95, kt 0..15) -> wih_p
__global__ void rcnn_pack_weights(const float* __restrict__ W_ih,
                                  const float* __restrict__ W_hh,
                                  unsigned int* __restrict__ whh_p,
                                  unsigned int* __restrict__ wih_p) {
    int bid  = blockIdx.x;
    int lane = threadIdx.x;
    const float* W;
    unsigned int* dst;
    int idx;
    if (bid < 512) { W = W_hh; dst = whh_p; idx = bid; }
    else           { W = W_ih; dst = wih_p; idx = bid - 512; }
    int nt = idx >> 4, kt = idx & 15;
    int n     = nt * 16 + (lane & 15);
    int kbase = kt * 32 + ((lane & 16) ? 16 : 0);
    unsigned int* out = dst + (size_t)idx * 256 + lane * 8;
    const float* row = W + (size_t)n * 512 + kbase;
#pragma unroll
    for (int v = 0; v < 8; ++v)
        out[v] = packbf(row[2 * v], row[2 * v + 1]);
}

// ---- fused scan ----------------------------------------------------------
// 8 workgroups (one per 16-row batch slab), 1024 threads = 32 waves.
// Wave w owns output columns [16w, 16w+16). Per step:
//   stage x_t slab (16x512 f32 -> bf16) into LDS, then per k-tile:
//   3 WMMAs for ih (wl/w1/w2 column groups) + 1 WMMA for h@W_hh^T,
//   then the elementwise recurrence directly on the C/D fragments.
__global__ __launch_bounds__(1024)
void rcnn_scan(const float* __restrict__ x,
               const int*   __restrict__ lengths,
               const float* __restrict__ b_hh,
               const float* __restrict__ bias,
               const unsigned int* __restrict__ whh_p,
               const unsigned int* __restrict__ wih_p,
               float* __restrict__ out) {
    __shared__ __attribute__((aligned(16))) unsigned short xs[16 * 512];
    __shared__ __attribute__((aligned(16))) unsigned short hs[16 * 512];

    const int tid  = threadIdx.x;
    const int w    = tid >> 5;
    const int lane = tid & 31;
    const int half = lane >> 4;
    const int ln   = lane & 15;
    const int b0   = blockIdx.x * 16;
    const int col  = w * 16 + ln;

    const float biasv = bias[col];
    const float bhhv  = b_hh[col];

    int lenv[8];
#pragma unroll
    for (int r = 0; r < 8; ++r) lenv[r] = lengths[b0 + half * 8 + r];

    float h[8], c1[8], c2[8];
#pragma unroll
    for (int r = 0; r < 8; ++r) { h[r] = 0.f; c1[r] = 0.f; c2[r] = 0.f; }

    for (int i = tid; i < 16 * 512; i += 1024) hs[i] = 0;
    __syncthreads();

    const int srow = tid >> 6;          // staging row 0..15
    const int sd0  = (tid & 63) * 8;    // staging col 0..511 step 8
    const float* xrow = x + ((size_t)(b0 + srow) * Tv) * Dv + sd0;

#pragma unroll 1
    for (int t = 0; t < Tv; ++t) {
        // Launder a zero dword-offset each step: defeats LICM on the weight
        // fragment loads (so they stream from L2 instead of being hoisted and
        // spilled) while keeping the pointers kernel-arg-derived so the
        // backend still selects global_load_b128 (LOADcnt only, no FLAT/DScnt
        // coupling with the LDS A-fragment traffic).
        int zoff = 0;
        asm volatile("" : "+v"(zoff));

        // B-fragment bases for this wave (each tile is 256 dwords).
        const unsigned int* whhB  = whh_p + (size_t)w * 4096 + lane * 8 + zoff;         // nt = w
        const unsigned int* wihB0 = wih_p + (size_t)(w)      * 4096 + lane * 8 + zoff;  // wl
        const unsigned int* wihB1 = wih_p + (size_t)(32 + w) * 4096 + lane * 8 + zoff;  // w1
        const unsigned int* wihB2 = wih_p + (size_t)(64 + w) * 4096 + lane * 8 + zoff;  // w2

        // ---- stage x_t slab into LDS as bf16 -----------------------------
        {
            const float4* xp = (const float4*)(xrow + (size_t)t * Dv);
            float4 a = xp[0], b = xp[1];
            uint4 pk = make_uint4(packbf(a.x, a.y), packbf(a.z, a.w),
                                  packbf(b.x, b.y), packbf(b.z, b.w));
            *(uint4*)(xs + srow * 512 + sd0) = pk;
            // Prefetch next step's slab (speculative, hides HBM/L2 latency).
            int tn = (t + 1 < Tv) ? (t + 1) : t;
            __builtin_prefetch(xrow + (size_t)tn * Dv, 0, 1);
        }
        __syncthreads();

        // ---- GEMMs: 64 WMMAs per wave ------------------------------------
        v8f a_wl = {}, a_w1 = {}, a_w2 = {}, a_hh = {};
#pragma unroll
        for (int kt = 0; kt < 16; ++kt) {
            const int k0 = kt * 32;
            v16bf ax = load_a_lds(xs, ln, k0, half);
            v16bf ah = load_a_lds(hs, ln, k0, half);
            a_wl = wmma_bf16(ax, load_b_glb(wihB0 + kt * 256), a_wl);
            a_w1 = wmma_bf16(ax, load_b_glb(wihB1 + kt * 256), a_w1);
            a_w2 = wmma_bf16(ax, load_b_glb(wihB2 + kt * 256), a_w2);
            a_hh = wmma_bf16(ah, load_b_glb(whhB  + kt * 256), a_hh);
        }
        __syncthreads();   // all xs/hs reads done before rewrite

        // ---- elementwise recurrence on fragments -------------------------
#pragma unroll
        for (int r = 0; r < 8; ++r) {
            float z   = a_wl[r] + a_hh[r] + bhhv;
            float lam = 1.0f / (1.0f + __expf(-z));
            float nc1 = lam * c1[r] + (1.0f - lam) * a_w1[r];
            float nc2 = lam * c2[r] + (1.0f - lam) * (c1[r] + a_w2[r]);
            float nh  = tanhf(nc2 + biasv);
            bool  m   = t < lenv[r];
            h[r]  = m ? nh  : h[r];
            c1[r] = m ? nc1 : c1[r];
            c2[r] = m ? nc2 : c2[r];
            int brow = half * 8 + r;
            out[((size_t)(b0 + brow) * Tv + t) * Hv + col] = m ? nh : 0.0f;
            hs[brow * 512 + col] = f2bf(h[r]);
        }
        // (next iteration's staging barrier orders hs writes vs. reads)
    }

    // ---- final state: h frozen after each row's last valid step ----------
#pragma unroll
    for (int r = 0; r < 8; ++r) {
        int brow = half * 8 + r;
        out[(size_t)Bv * Tv * Hv + (size_t)(b0 + brow) * Hv + col] = h[r];
    }
}

// ---- launcher ------------------------------------------------------------
extern "C" void kernel_launch(void* const* d_in, const int* in_sizes, int n_in,
                              void* d_out, int out_size, void* d_ws, size_t ws_size,
                              hipStream_t stream) {
    const float* x       = (const float*)d_in[0];
    // d_in[1] is the bool mask; we reconstruct it from lengths instead.
    const int*   lengths = (const int*)d_in[2];
    const float* W_ih    = (const float*)d_in[3];
    const float* W_hh    = (const float*)d_in[4];
    const float* b_hh    = (const float*)d_in[5];
    const float* bias    = (const float*)d_in[6];
    float*       out     = (float*)d_out;

    unsigned int* whh_p = (unsigned int*)d_ws;            // 512 KB
    unsigned int* wih_p = whh_p + 512 * 256;              // 1.5 MB

    rcnn_pack_weights<<<2048, 32, 0, stream>>>(W_ih, W_hh, whh_p, wih_p);
    rcnn_scan<<<Bv / 16, 1024, 0, stream>>>(x, lengths, b_hh, bias,
                                            whh_p, wih_p, out);
}
